// MultiHeadAttention_47811575939777
// MI455X (gfx1250) — matrix operand order repaired and checked by reading, not compiled, not run on hardware
//
#include <hip/hip_runtime.h>
#include <hip/hip_bf16.h>

// Problem constants (from reference): B=4, C=64, D=128, CV=64, HV=WV=64
#define NB   4
#define NC   64
#define ND   128
#define NCV  64
#define NHW  4096              // HV*WV
#define FEAT_ELEMS (NB * NC * NHW)   // 1,048,576 floats

typedef __attribute__((ext_vector_type(16))) _Float16 v16h;
typedef __attribute__((ext_vector_type(8)))  float    v8f;

// -----------------------------------------------------------------------------
// Kernel 1: attention scores via WMMA, faithful singleton softmax -> att
// One wave32 per 16x16 output tile; EXEC all-ones (WMMA requirement).
// scores[b,i,j] = scale * sum_d q[b,i,d]*k[b,j,d]; att = softmax over axis-3
// (singleton) = exp(x-x)/exp(x-x) == 1.0 for all finite x, computed faithfully.
// -----------------------------------------------------------------------------
__global__ __launch_bounds__(32)
void att_wmma_kernel(const float* __restrict__ q,
                     const float* __restrict__ k,
                     float* __restrict__ att) {
  const int b    = blockIdx.x >> 4;        // 0..3
  const int ti   = (blockIdx.x >> 2) & 3;  // i-tile 0..3
  const int tj   = blockIdx.x & 3;         // j-tile 0..3
  const int lane = threadIdx.x;            // 0..31 (wave32)
  const int half = lane >> 4;
  const int l16  = lane & 15;

  const float* qrow = q + (size_t)((b * NC) + ti * 16 + l16) * ND;
  const float* krow = k + (size_t)((b * NC) + tj * 16 + l16) * ND;

  v8f acc = {};
  #pragma unroll
  for (int kk = 0; kk < 4; ++kk) {         // K = 128 = 4 x 32
    const int kbase = kk * 32;
    v16h a, bm;
    // A (16x32 f16): lane half selects K+8; VGPR r covers K = {2r,2r+1} (+16 for r>=4)
    // B (32x16 f16): same K order as A (lane half selects K+8; c>=8 adds +8)
    #pragma unroll
    for (int c = 0; c < 16; ++c) {
      const int r  = c >> 1;
      const int Ka = ((r < 4) ? 0 : 16) + 2 * (r & 3) + (c & 1) + 8 * half;
      const int Kb = ((c < 8) ? 0 : 8) + c + 8 * half;
      a[c]  = (_Float16)qrow[kbase + Ka];
      bm[c] = (_Float16)krow[kbase + Kb];
    }
    acc = __builtin_amdgcn_wmma_f32_16x16x32_f16(
        /*neg_a=*/false, a, /*neg_b=*/false, bm,
        /*c_mod=*/(short)0, acc, /*reuse_a=*/false, /*reuse_b=*/false);
  }

  const float scale = 0.08838834764831845f;  // 128^-0.5
  // C/D layout: VGPR c holds rows M=c (lanes 0-15) and M=c+8 (lanes 16-31)
  #pragma unroll
  for (int c = 0; c < 8; ++c) {
    const int i = ti * 16 + c + 8 * half;
    const int j = tj * 16 + l16;
    float x = acc[c] * scale;
    float m = x;                  // max over singleton softmax axis
    float e = __expf(x - m);      // numerator
    float s = e;                  // denominator (sum over singleton axis)
    att[((size_t)(b * NC) + i) * NC + j] = e / s;   // == 1.0 for finite x
  }
}

// -----------------------------------------------------------------------------
// Kernel 2 (bandwidth-bound, 256 MB stream): per (b,cv) block reduce over j.
// partial[b,cv,n] = w1[cv] * sum_j w2[j] * v[b,cv,j,n]
// -----------------------------------------------------------------------------
__device__ __forceinline__ void fma4(float4& a, float w, const float4 x) {
  a.x += w * x.x; a.y += w * x.y; a.z += w * x.z; a.w += w * x.w;
}

__global__ __launch_bounds__(256)
void vw_partial_kernel(const float* __restrict__ v,
                       const float* __restrict__ w1,
                       const float* __restrict__ w2,
                       float* __restrict__ partial) {
  const int b  = blockIdx.x >> 6;    // 0..3
  const int cv = blockIdx.x & 63;    // 0..63
  const int t  = threadIdx.x;        // 0..255, each owns 4x float4 across n

  const float* vbase = v + (size_t)(b * NCV + cv) * NC * NHW;

  float4 a0 = {0,0,0,0}, a1 = {0,0,0,0}, a2 = {0,0,0,0}, a3 = {0,0,0,0};
  for (int j = 0; j < NC; ++j) {
    const float wj = w2[j];
    const float4* row = reinterpret_cast<const float4*>(vbase + (size_t)j * NHW) + t;
    if (j + 1 < NC) {
      // pull next row's cachelines early (global_prefetch_b8)
      __builtin_prefetch(vbase + (size_t)(j + 1) * NHW + t * 4, 0, 0);
    }
    float4 x0 = row[0];
    float4 x1 = row[256];
    float4 x2 = row[512];
    float4 x3 = row[768];
    fma4(a0, wj, x0); fma4(a1, wj, x1); fma4(a2, wj, x2); fma4(a3, wj, x3);
  }
  const float wc = w1[cv];
  a0.x *= wc; a0.y *= wc; a0.z *= wc; a0.w *= wc;
  a1.x *= wc; a1.y *= wc; a1.z *= wc; a1.w *= wc;
  a2.x *= wc; a2.y *= wc; a2.z *= wc; a2.w *= wc;
  a3.x *= wc; a3.y *= wc; a3.z *= wc; a3.w *= wc;

  float4* p = reinterpret_cast<float4*>(partial + (size_t)(b * NCV + cv) * NHW) + t;
  p[0] = a0; p[256] = a1; p[512] = a2; p[768] = a3;
}

// -----------------------------------------------------------------------------
// Kernel 3: reduce partials over cv, add bias constant, broadcast over i.
// feature[b,i,n] = sum_cv partial[b,cv,n] + b1*sum(w2) + b2   (same for all i)
// -----------------------------------------------------------------------------
__global__ __launch_bounds__(256)
void feature_kernel(const float* __restrict__ partial,
                    const float* __restrict__ w2,
                    const float* __restrict__ b1,
                    const float* __restrict__ b2,
                    float* __restrict__ feature) {
  const int b = blockIdx.x >> 4;                     // 0..3
  const int n = (blockIdx.x & 15) * 256 + threadIdx.x;  // 0..4095

  float sw2 = 0.0f;
  #pragma unroll
  for (int j = 0; j < NC; ++j) sw2 += w2[j];         // uniform scalar loads
  const float cst = b1[0] * sw2 + b2[0];

  float s = cst;
  const float* p = partial + (size_t)b * NCV * NHW + n;
  #pragma unroll 8
  for (int cv = 0; cv < NCV; ++cv) s += p[(size_t)cv * NHW];

  float* f = feature + (size_t)b * NC * NHW + n;
  #pragma unroll 8
  for (int i = 0; i < NC; ++i) f[(size_t)i * NHW] = s;
}

// -----------------------------------------------------------------------------
extern "C" void kernel_launch(void* const* d_in, const int* in_sizes, int n_in,
                              void* d_out, int out_size, void* d_ws, size_t ws_size,
                              hipStream_t stream) {
  (void)in_sizes; (void)n_in; (void)out_size; (void)ws_size;
  const float* q  = (const float*)d_in[0];
  const float* k  = (const float*)d_in[1];
  const float* v  = (const float*)d_in[2];
  const float* w1 = (const float*)d_in[3];
  const float* b1 = (const float*)d_in[4];
  const float* w2 = (const float*)d_in[5];
  const float* b2 = (const float*)d_in[6];

  float* feature = (float*)d_out;                 // [B, C, HV, WV]
  float* att     = feature + FEAT_ELEMS;          // [B, C, C, 1]
  float* partial = (float*)d_ws;                  // [B, CV, HW] = 4 MB scratch

  // scores via WMMA + faithful singleton softmax -> att
  att_wmma_kernel<<<NB * 4 * 4, 32, 0, stream>>>(q, k, att);
  // 256 MB streaming reduction over j (the roofline-dominant kernel)
  vw_partial_kernel<<<NB * NCV, 256, 0, stream>>>(v, w1, w2, partial);
  // reduce over cv + broadcast over channels
  feature_kernel<<<NB * 16, 256, 0, stream>>>(partial, w2, b1, b2, feature);
}
